// CausalSelfAttention_69329362092472
// MI455X (gfx1250) — matrix operand order repaired and checked by reading, not compiled, not run on hardware
//
#include <hip/hip_runtime.h>
#include <hip/hip_bf16.h>

// ---------------------------------------------------------------------------
// Causal self-attention, MI455X (gfx1250, wave32, WMMA).
// Shapes (compile-time): B=2, L=2048, D=1024, H=16, dh=64.
// bf16 WMMA (v_wmma_f32_16x16x32_bf16) with fp32 accumulation for all three
// matmul stages; async global->LDS staging where no transpose is needed.
// ---------------------------------------------------------------------------

typedef __attribute__((ext_vector_type(16))) __bf16 v16bf;
typedef __attribute__((ext_vector_type(8)))  float  v8f;

constexpr int Bsz  = 2;
constexpr int Lseq = 2048;
constexpr int Dm   = 1024;
constexpr int Hh   = 16;
constexpr int Dh   = 64;            // Dm / Hh
constexpr int Mtot = Bsz * Lseq;    // 4096 rows
constexpr int ThreeD = 3 * Dm;      // 3072

union Frag {
    v16bf v;
    uint4 u[2];
    unsigned short s[16];
};

__device__ __forceinline__ unsigned short f2bf(float f) {
    unsigned int u = __float_as_uint(f);
    u += 0x7FFFu + ((u >> 16) & 1u);   // round to nearest even
    return (unsigned short)(u >> 16);
}

// ---------------------------------------------------------------------------
// Async global->LDS staging (CDNA5 GLOBAL_LOAD_ASYNC_TO_LDS_B128, ASYNCcnt).
// Builtin signature (from hipcc diagnostic): (v4i AS1*, v4i AS3*, Ii, Ii).
// Guarded so the file still compiles (sync fallback) if the builtin is absent.
// ---------------------------------------------------------------------------
typedef int v4i __attribute__((vector_size(16)));
typedef __attribute__((address_space(1))) v4i* gv4i_p;   // global int4*
typedef __attribute__((address_space(3))) v4i* lv4i_p;   // LDS int4*

#if __has_builtin(__builtin_amdgcn_global_load_async_to_lds_b128)
#define HAVE_ASYNC_LDS 1
__device__ __forceinline__ void async_copy_b128(void* lds_dst, const void* g_src) {
    __builtin_amdgcn_global_load_async_to_lds_b128(
        (gv4i_p)g_src, (lv4i_p)lds_dst, 0, 0);
}
#else
#define HAVE_ASYNC_LDS 0
__device__ __forceinline__ void async_copy_b128(void* lds_dst, const void* g_src) {
    *(uint4*)lds_dst = *(const uint4*)g_src;
}
#endif

__device__ __forceinline__ void wait_async_lds() {
#if HAVE_ASYNC_LDS
#if __has_builtin(__builtin_amdgcn_s_wait_asynccnt)
    __builtin_amdgcn_s_wait_asynccnt(0);
#else
    asm volatile("s_wait_asynccnt 0" ::: "memory");
#endif
#endif
}

// ---------------------------------------------------------------------------
// Kernel: fp32 -> bf16 cast, 8 elements / thread (b128 in, b128 out)
// ---------------------------------------------------------------------------
__global__ void cast_f32_to_bf16_vec8(const float* __restrict__ in,
                                      unsigned short* __restrict__ out,
                                      int n8) {
    int i = blockIdx.x * blockDim.x + threadIdx.x;
    if (i >= n8) return;
    const float4* in4 = (const float4*)in;
    float4 a = in4[2 * i];
    float4 b = in4[2 * i + 1];
    union { uint4 u; unsigned short s[8]; } r;
    r.s[0] = f2bf(a.x); r.s[1] = f2bf(a.y); r.s[2] = f2bf(a.z); r.s[3] = f2bf(a.w);
    r.s[4] = f2bf(b.x); r.s[5] = f2bf(b.y); r.s[6] = f2bf(b.z); r.s[7] = f2bf(b.w);
    ((uint4*)out)[i] = r.u;
}

// ---------------------------------------------------------------------------
// Kernel: C[M,N] = A[M,K] * W[K,N], bf16 inputs, fp32 accumulate via WMMA.
// Block tile 128x128, K-step 32, 256 threads = 8 waves (2x4 wave grid),
// each wave does a 64x32 sub-tile = 4x2 WMMA tiles.
// A tile staged with async global->LDS (row-major, no transpose needed).
// W tile staged TRANSPOSED [col][k] (sync, via registers) so B-fragments are
// contiguous b128 LDS reads matching the ISA layouts:
//   A (16-bit 16x32): lanes 0-15 row=l, K 0..7 & 16..23; lanes 16-31 K 8..15 & 24..31
//   B (16-bit 32x16): lanes 0-15 col=l, K 0..7 & 16..23;        lanes 16-31 K 8..15 & 24..31
// ---------------------------------------------------------------------------
template<bool OUT_BF16>
__global__ __launch_bounds__(256)
void gemm_bf16_wmma(const unsigned short* __restrict__ A,
                    const unsigned short* __restrict__ W,
                    float* __restrict__ Cf,
                    unsigned short* __restrict__ Cb,
                    int M, int N, int K) {
    constexpr int BM = 128, BN = 128, BK = 32;
    constexpr int LDA = BK + 8;   // 40 (pad to dodge bank conflicts)
    constexpr int LDB = BK + 8;   // 40
    __shared__ unsigned short As[BM * LDA];
    __shared__ unsigned short Wt[BN * LDB];

    const int tid  = threadIdx.x;
    const int wave = tid >> 5;
    const int lane = tid & 31;
    const int lh   = lane & 15;
    const int hi   = lane >> 4;          // 0: lanes 0-15, 1: lanes 16-31
    const int wm   = wave >> 2;          // 0..1
    const int wn   = wave & 3;           // 0..3
    const int m0   = blockIdx.y * BM;
    const int n0   = blockIdx.x * BN;

    v8f acc[4][2];
#pragma unroll
    for (int i = 0; i < 4; ++i)
#pragma unroll
        for (int j = 0; j < 2; ++j)
            acc[i][j] = (v8f){0.f, 0.f, 0.f, 0.f, 0.f, 0.f, 0.f, 0.f};

    for (int k0 = 0; k0 < K; k0 += BK) {
        // ---- stage A tile (async): 128x32 bf16, 512 b128 chunks, 2/thread --
#pragma unroll
        for (int cc = 0; cc < 2; ++cc) {
            int c    = tid + cc * 256;
            int row  = c >> 2;
            int col8 = (c & 3) * 8;
            async_copy_b128(As + row * LDA + col8,
                            A + (size_t)(m0 + row) * K + k0 + col8);
        }
        // ---- stage W tile transposed (sync): Wt[n][k] ----
#pragma unroll
        for (int cc = 0; cc < 2; ++cc) {
            int c    = tid + cc * 256;
            int row  = c >> 4;           // k within tile, 0..31
            int col8 = (c & 15) * 8;     // n within tile
            union { uint4 u; unsigned short s[8]; } val;
            val.u = *(const uint4*)(W + (size_t)(k0 + row) * N + n0 + col8);
#pragma unroll
            for (int e = 0; e < 8; ++e)
                Wt[(col8 + e) * LDB + row] = val.s[e];
        }
        if (k0 + BK < K)   // prefetch next W tile row for this thread
            __builtin_prefetch(W + (size_t)(k0 + BK + (tid >> 4)) * N + n0, 0, 0);
        wait_async_lds();
        __syncthreads();

        // ---- fragments from LDS ----
        Frag a[4], b[2];
#pragma unroll
        for (int i = 0; i < 4; ++i) {
            const unsigned short* p = As + (wm * 64 + i * 16 + lh) * LDA + hi * 8;
            a[i].u[0] = *(const uint4*)p;
            a[i].u[1] = *(const uint4*)(p + 16);
        }
#pragma unroll
        for (int j = 0; j < 2; ++j) {
            const unsigned short* p = Wt + (wn * 32 + j * 16 + lh) * LDB + hi * 8;
            b[j].u[0] = *(const uint4*)p;
            b[j].u[1] = *(const uint4*)(p + 16);
        }
#pragma unroll
        for (int i = 0; i < 4; ++i)
#pragma unroll
            for (int j = 0; j < 2; ++j)
                acc[i][j] = __builtin_amdgcn_wmma_f32_16x16x32_bf16(
                    false, a[i].v, false, b[j].v, (short)0, acc[i][j], false, false);
        __syncthreads();
    }

    // ---- store: C layout lane(l,VGPR r) -> row r + 8*hi, col lh ----
#pragma unroll
    for (int i = 0; i < 4; ++i) {
#pragma unroll
        for (int j = 0; j < 2; ++j) {
            int colg = n0 + wn * 32 + j * 16 + lh;
#pragma unroll
            for (int r = 0; r < 8; ++r) {
                int rowg = m0 + wm * 64 + i * 16 + r + hi * 8;
                float v = acc[i][j][r];
                if (OUT_BF16) Cb[(size_t)rowg * N + colg] = f2bf(v);
                else          Cf[(size_t)rowg * N + colg] = v;
            }
        }
    }
}

// ---------------------------------------------------------------------------
// Flash attention kernel.
// qkv: [B, L, 3D] bf16 (Q at col h*64, K at D + h*64, V at 2D + h*64)
// o:   [B, L, D] bf16
// Grid: (L/128, H, B); 256 threads = 8 waves; wave w owns Q rows
// q_base..q_base+15. Per 32-key step the whole block stages the shared
// K tile (async, Kt[key][dh]) and transposed V tile (sync, Vt[dh][key]).
// ---------------------------------------------------------------------------
__global__ __launch_bounds__(256)
void flash_attn_wmma(const unsigned short* __restrict__ qkv,
                     unsigned short* __restrict__ o) {
    constexpr int LDK = Dh + 8;   // 72, Kt [32 keys][64 dh]
    constexpr int LDV = 32 + 8;   // 40, Vt [64 dh][32 keys]
    constexpr int LDP = 32 + 8;   // 40, Pt [16 rows][32 keys] per wave
    __shared__ unsigned short Kt[32 * LDK];
    __shared__ unsigned short Vt[64 * LDV];
    __shared__ unsigned short Pt[8][16 * LDP];

    const int tid  = threadIdx.x;
    const int wave = tid >> 5;
    const int lane = tid & 31;
    const int lh   = lane & 15;
    const int hi   = lane >> 4;
    const int bb   = blockIdx.z;
    const int h    = blockIdx.y;
    const int qb0  = blockIdx.x * 128;
    const int q_base = qb0 + wave * 16;

    const size_t rs = (size_t)ThreeD;                       // qkv row stride
    const unsigned short* qbase = qkv + (size_t)bb * Lseq * rs + h * Dh;
    const unsigned short* kbase = qbase + Dm;
    const unsigned short* vbase = qbase + 2 * Dm;

    // ---- persistent Q fragments (A operand, K=dh split into 2x32) ----
    Frag aQ[2];
    {
        const unsigned short* p = qbase + (size_t)(q_base + lh) * rs;
#pragma unroll
        for (int kk = 0; kk < 2; ++kk) {
            aQ[kk].u[0] = *(const uint4*)(p + kk * 32 + hi * 8);
            aQ[kk].u[1] = *(const uint4*)(p + kk * 32 + hi * 8 + 16);
        }
    }

    v8f accO[4];
#pragma unroll
    for (int t = 0; t < 4; ++t)
        accO[t] = (v8f){0.f, 0.f, 0.f, 0.f, 0.f, 0.f, 0.f, 0.f};
    float mrow[8], srow[8];
#pragma unroll
    for (int r = 0; r < 8; ++r) { mrow[r] = -1.0e30f; srow[r] = 0.f; }

    const float scale = 0.125f;   // 1/sqrt(64)
    const int jend = qb0 + 128;   // block-uniform trip count (causal bound)

    for (int j = 0; j < jend; j += 32) {
        __syncthreads();
        // ---- stage K tile (async): Kt[key][dh], 32x64, one b128/thread ----
        {
            int row = tid >> 3, c8 = (tid & 7) * 8;
            async_copy_b128(Kt + row * LDK + c8,
                            kbase + (size_t)(j + row) * rs + c8);
        }
        // ---- stage V tile transposed (sync): Vt[dh][key] ----
        {
            int row = tid >> 3, c8 = (tid & 7) * 8;
            union { uint4 u; unsigned short s[8]; } val;
            val.u = *(const uint4*)(vbase + (size_t)(j + row) * rs + c8);
#pragma unroll
            for (int e = 0; e < 8; ++e)
                Vt[(c8 + e) * LDV + row] = val.s[e];
        }
        wait_async_lds();
        __syncthreads();

        if (j > q_base + 15) continue;   // wave-uniform: fully masked tile

        // ---- S = (Q * K^T) for keys j..j+31, two 16x16 C tiles ----
        v8f S[2];
#pragma unroll
        for (int t = 0; t < 2; ++t) {
            v8f c = (v8f){0.f, 0.f, 0.f, 0.f, 0.f, 0.f, 0.f, 0.f};
#pragma unroll
            for (int kk = 0; kk < 2; ++kk) {
                Frag bK;
                const unsigned short* p = Kt + (t * 16 + lh) * LDK + kk * 32 + hi * 8;
                bK.u[0] = *(const uint4*)p;
                bK.u[1] = *(const uint4*)(p + 16);
                c = __builtin_amdgcn_wmma_f32_16x16x32_bf16(
                    false, aQ[kk].v, false, bK.v, (short)0, c, false, false);
            }
            S[t] = c;
        }

        // ---- scale + causal mask (row q = q_base + r + 8*hi, key = j+16t+lh)
        const bool need_mask = (j + 31 > q_base);
#pragma unroll
        for (int t = 0; t < 2; ++t)
#pragma unroll
            for (int r = 0; r < 8; ++r) {
                float s = S[t][r] * scale;
                if (need_mask) {
                    int q   = q_base + r + hi * 8;
                    int key = j + t * 16 + lh;
                    if (key > q) s = -1.0e30f;
                }
                S[t][r] = s;
            }

        // ---- online softmax: row reductions across each 16-lane half ----
        float mloc[8];
#pragma unroll
        for (int r = 0; r < 8; ++r) mloc[r] = fmaxf(S[0][r], S[1][r]);
#pragma unroll
        for (int off = 8; off >= 1; off >>= 1)
#pragma unroll
            for (int r = 0; r < 8; ++r)
                mloc[r] = fmaxf(mloc[r], __shfl_xor(mloc[r], off, 32));

        float alpha[8], psum[8];
#pragma unroll
        for (int r = 0; r < 8; ++r) {
            float mnew = fmaxf(mrow[r], mloc[r]);
            alpha[r] = __expf(mrow[r] - mnew);
            mrow[r]  = mnew;
            psum[r]  = 0.f;
        }
        // P = exp(S - m); spill to per-wave LDS in A-matrix addressing order
#pragma unroll
        for (int t = 0; t < 2; ++t)
#pragma unroll
            for (int r = 0; r < 8; ++r) {
                float p = __expf(S[t][r] - mrow[r]);
                psum[r] += p;
                Pt[wave][(r + hi * 8) * LDP + t * 16 + lh] = f2bf(p);
            }
#pragma unroll
        for (int off = 8; off >= 1; off >>= 1)
#pragma unroll
            for (int r = 0; r < 8; ++r)
                psum[r] += __shfl_xor(psum[r], off, 32);
#pragma unroll
        for (int r = 0; r < 8; ++r)
            srow[r] = srow[r] * alpha[r] + psum[r];
#pragma unroll
        for (int t = 0; t < 4; ++t)
#pragma unroll
            for (int r = 0; r < 8; ++r)
                accO[t][r] *= alpha[r];

        // ---- O += P * V (A-frag from Pt, B-frags from Vt) ----
        Frag aP;
        {
            const unsigned short* pp = Pt[wave] + lh * LDP + hi * 8;
            aP.u[0] = *(const uint4*)pp;
            aP.u[1] = *(const uint4*)(pp + 16);
        }
#pragma unroll
        for (int t = 0; t < 4; ++t) {
            Frag bV;
            const unsigned short* p = Vt + (t * 16 + lh) * LDV + hi * 8;
            bV.u[0] = *(const uint4*)p;
            bV.u[1] = *(const uint4*)(p + 16);
            accO[t] = __builtin_amdgcn_wmma_f32_16x16x32_bf16(
                false, aP.v, false, bV.v, (short)0, accO[t], false, false);
        }
    }

    // ---- epilogue: normalize and store bf16 o[b, q, h*64 + dh] ----
    float inv[8];
#pragma unroll
    for (int r = 0; r < 8; ++r) inv[r] = 1.f / srow[r];
#pragma unroll
    for (int t = 0; t < 4; ++t)
#pragma unroll
        for (int r = 0; r < 8; ++r) {
            int q = q_base + r + hi * 8;
            o[((size_t)bb * Lseq + q) * Dm + h * Dh + t * 16 + lh] =
                f2bf(accO[t][r] * inv[r]);
        }
}

// ---------------------------------------------------------------------------
// Host launcher. Workspace layout (ushort elements), total ~48 MB:
//   x_bf[4096*1024] wqkv_bf[1024*3072] wproj_bf[1024*1024]
//   qkv_bf[4096*3072] o_bf[4096*1024]
// ---------------------------------------------------------------------------
extern "C" void kernel_launch(void* const* d_in, const int* in_sizes, int n_in,
                              void* d_out, int out_size, void* d_ws, size_t ws_size,
                              hipStream_t stream) {
    (void)in_sizes; (void)n_in; (void)out_size; (void)ws_size;
    const float* x      = (const float*)d_in[0];
    const float* w_qkv  = (const float*)d_in[1];
    const float* w_proj = (const float*)d_in[2];
    float* out = (float*)d_out;

    unsigned short* x_bf     = (unsigned short*)d_ws;
    unsigned short* wqkv_bf  = x_bf     + (size_t)Mtot * Dm;
    unsigned short* wproj_bf = wqkv_bf  + (size_t)Dm * ThreeD;
    unsigned short* qkv_bf   = wproj_bf + (size_t)Dm * Dm;
    unsigned short* o_bf     = qkv_bf   + (size_t)Mtot * ThreeD;

    // 1) casts
    {
        int n8 = Mtot * Dm / 8;
        cast_f32_to_bf16_vec8<<<(n8 + 255) / 256, 256, 0, stream>>>(x, x_bf, n8);
        n8 = Dm * ThreeD / 8;
        cast_f32_to_bf16_vec8<<<(n8 + 255) / 256, 256, 0, stream>>>(w_qkv, wqkv_bf, n8);
        n8 = Dm * Dm / 8;
        cast_f32_to_bf16_vec8<<<(n8 + 255) / 256, 256, 0, stream>>>(w_proj, wproj_bf, n8);
    }
    // 2) qkv = x @ w_qkv   (bf16 out)
    gemm_bf16_wmma<true><<<dim3(ThreeD / 128, Mtot / 128), 256, 0, stream>>>(
        x_bf, wqkv_bf, nullptr, qkv_bf, Mtot, ThreeD, Dm);
    // 3) flash attention -> o_bf [B, L, D]
    flash_attn_wmma<<<dim3(Lseq / 128, Hh, Bsz), 256, 0, stream>>>(qkv_bf, o_bf);
    // 4) out = o @ w_proj  (fp32 out)
    gemm_bf16_wmma<false><<<dim3(Dm / 128, Mtot / 128), 256, 0, stream>>>(
        o_bf, wproj_bf, out, nullptr, Mtot, Dm, Dm);
}
